// GNNFullyConnected_8134668059259
// MI455X (gfx1250) — matrix operand order repaired and checked by reading, not compiled, not run on hardware
//
#include <hip/hip_runtime.h>
#include <math.h>

typedef __attribute__((ext_vector_type(16))) _Float16 v16h;
typedef __attribute__((ext_vector_type(8)))  float    v8f;

#define G_   64
#define N_   1024
#define DIN  128
#define DH   256
#define DOUT 10

// ---------------- prep: W1 (f32 [128,256]) -> transposed f16 hi/lo [256,128]
__global__ void k_prep_w1(const float* __restrict__ W1,
                          _Float16* __restrict__ w1t_hi,
                          _Float16* __restrict__ w1t_lo) {
  int k = blockIdx.x;     // 0..127 (D_IN)
  int n = threadIdx.x;    // 0..255 (D_H)
  float v = W1[k * DH + n];
  _Float16 hi = (_Float16)v;
  _Float16 lo = (_Float16)(v - (float)hi);
  w1t_hi[n * DIN + k] = hi;
  w1t_lo[n * DIN + k] = lo;
}

// ---------------- per-node reciprocal norm (one wave32 per node)
__global__ void k_rnorm(const float* __restrict__ x, float* __restrict__ rnorm) {
  int node = (blockIdx.x * blockDim.x + threadIdx.x) >> 5;
  int lane = threadIdx.x & 31;
  const float* row = x + (size_t)node * DIN;
  float s = 0.f;
#pragma unroll
  for (int i = 0; i < 4; ++i) { float v = row[lane + 32 * i]; s += v * v; }
  for (int off = 16; off > 0; off >>= 1) s += __shfl_xor(s, off, 32);
  if (lane == 0) rnorm[node] = rsqrtf(s + 1e-24f);
}

// ---------------- per-graph c[g,d] = sum_n xn[g,n,d]   (64 blocks x 128 thr)
__global__ void k_csum(const float* __restrict__ x, const float* __restrict__ rnorm,
                       float* __restrict__ c) {
  int g = blockIdx.x, d = threadIdx.x;
  const float* xg = x + (size_t)g * N_ * DIN;
  const float* rg = rnorm + g * N_;
  float acc = 0.f;
  for (int n = 0; n < N_; ++n) acc += xg[(size_t)n * DIN + d] * rg[n];
  c[g * DIN + d] = acc;
}

// ---------------- per-node pooling weight w[m] = c_g . xn[m] - 1
__global__ void k_wcoef(const float* __restrict__ x, const float* __restrict__ rnorm,
                        const float* __restrict__ c, float* __restrict__ w) {
  int node = (blockIdx.x * blockDim.x + threadIdx.x) >> 5;
  int lane = threadIdx.x & 31;
  int g = node >> 10;
  const float* row = x + (size_t)node * DIN;
  const float* cg  = c + g * DIN;
  float s = 0.f;
#pragma unroll
  for (int i = 0; i < 4; ++i) s += row[lane + 32 * i] * cg[lane + 32 * i];
  for (int off = 16; off > 0; off >>= 1) s += __shfl_xor(s, off, 32);
  if (lane == 0) w[node] = rnorm[node] * s - 1.0f;
}

// ---------------- fused fc1(+bias,+relu) via WMMA + weighted node pooling
// one wave per 16-node row tile; split-f16 (hi/lo) gives ~f32 accuracy.
__global__ __launch_bounds__(256) void k_fc1_pool(
    const float* __restrict__ x,
    const _Float16* __restrict__ w1t_hi, const _Float16* __restrict__ w1t_lo,
    const float* __restrict__ b1, const float* __restrict__ w,
    float* __restrict__ partial) {
  int waveId = (blockIdx.x * blockDim.x + threadIdx.x) >> 5;  // 0..4095
  int lane = threadIdx.x & 31;
  int half = lane >> 4;      // 0 or 1
  int l16  = lane & 15;
  int tileBase = waveId * 16;

  // A fragments (16x32 f16 layout): lane<16 -> K 0..15 ; lane>=16 -> K 16..31 (contiguous)
  v16h a_hi[4], a_lo[4];
  const float* rowp = x + (size_t)(tileBase + l16) * DIN;
#pragma unroll
  for (int kc = 0; kc < 4; ++kc) {
    int kb0 = kc * 32 + half * 16;
    int kb1 = kb0 + 8;
#pragma unroll
    for (int e = 0; e < 8; ++e) {
      float v0 = rowp[kb0 + e];
      _Float16 h0 = (_Float16)v0;
      a_hi[kc][e]     = h0;
      a_lo[kc][e]     = (_Float16)(v0 - (float)h0);
      float v1 = rowp[kb1 + e];
      _Float16 h1 = (_Float16)v1;
      a_hi[kc][8 + e] = h1;
      a_lo[kc][8 + e] = (_Float16)(v1 - (float)h1);
    }
  }
  // per-lane node pooling weights (C rows covered by this lane: base + v)
  float wv[8];
#pragma unroll
  for (int v = 0; v < 8; ++v) wv[v] = w[tileBase + half * 8 + v];

#pragma unroll 1
  for (int nt = 0; nt < 16; ++nt) {
    int col = nt * 16 + l16;
    float bias = b1[col];
    v8f acc;
#pragma unroll
    for (int v = 0; v < 8; ++v) acc[v] = bias;

#pragma unroll
    for (int kc = 0; kc < 4; ++kc) {
      // B fragment (32x16 f16): lane column = l16, K range = half*16 .. +15 (contiguous)
      int kbase = kc * 32 + half * 16;
      v16h bh = *reinterpret_cast<const v16h*>(w1t_hi + (size_t)col * DIN + kbase);
      v16h bl = *reinterpret_cast<const v16h*>(w1t_lo + (size_t)col * DIN + kbase);
      acc = __builtin_amdgcn_wmma_f32_16x16x32_f16(false, a_hi[kc], false, bh,
                                                   (short)0, acc, false, false);
      acc = __builtin_amdgcn_wmma_f32_16x16x32_f16(false, a_lo[kc], false, bh,
                                                   (short)0, acc, false, false);
      acc = __builtin_amdgcn_wmma_f32_16x16x32_f16(false, a_hi[kc], false, bl,
                                                   (short)0, acc, false, false);
    }
    // relu + weighted pool over this wave's 16 rows
    float pr = 0.f;
#pragma unroll
    for (int v = 0; v < 8; ++v) {
      float hval = acc[v] > 0.f ? acc[v] : 0.f;
      pr += wv[v] * hval;
    }
    pr += __shfl_xor(pr, 16, 32);           // combine rows 0-7 with rows 8-15
    if (half == 0) partial[(size_t)waveId * DH + col] = pr;
  }
}

// ---------------- deterministic reduction of per-wave partials
__global__ void k_reduce(const float* __restrict__ partial, float* __restrict__ pooled) {
  int g = blockIdx.x, cidx = threadIdx.x;   // 64 x 256
  float s = 0.f;
  for (int t = 0; t < 64; ++t) s += partial[((size_t)g * 64 + t) * DH + cidx];
  pooled[g * DH + cidx] = s;
}

// ---------------- head: mean-pool scale, fc2, log_softmax (one wave per graph)
__global__ void k_head(const float* __restrict__ pooled, const float* __restrict__ W2,
                       const float* __restrict__ b2, float* __restrict__ out) {
  int g = blockIdx.x, lane = threadIdx.x;   // 64 x 32
  float o = -INFINITY;
  if (lane < DOUT) {
    float acc = 0.f;
    for (int d = 0; d < DH; ++d) acc += pooled[g * DH + d] * W2[d * DOUT + lane];
    o = b2[lane] + acc * (1.0f / (float)N_);
  }
  float m = o;
  for (int off = 16; off > 0; off >>= 1) m = fmaxf(m, __shfl_xor(m, off, 32));
  float e = (lane < DOUT) ? expf(o - m) : 0.f;
  float s = e;
  for (int off = 16; off > 0; off >>= 1) s += __shfl_xor(s, off, 32);
  if (lane < DOUT) out[g * DOUT + lane] = (o - m) - logf(s);
}

extern "C" void kernel_launch(void* const* d_in, const int* in_sizes, int n_in,
                              void* d_out, int out_size, void* d_ws, size_t ws_size,
                              hipStream_t stream) {
  const float* x  = (const float*)d_in[0];
  // d_in[1] = batch (int64): unused, graphs are equal-sized and sorted
  const float* W1 = (const float*)d_in[2];
  const float* b1 = (const float*)d_in[3];
  const float* W2 = (const float*)d_in[4];
  const float* b2 = (const float*)d_in[5];
  float* out = (float*)d_out;

  char* ws = (char*)d_ws;
  _Float16* w1t_hi = (_Float16*)(ws + 0);        //  65536 B
  _Float16* w1t_lo = (_Float16*)(ws + 65536);    //  65536 B
  float* rnorm     = (float*)(ws + 131072);      // 262144 B
  float* c         = (float*)(ws + 393216);      //  32768 B
  float* w         = (float*)(ws + 425984);      // 262144 B
  float* partial   = (float*)(ws + 688128);      // 4 MiB (4096 waves x 256)
  float* pooled    = (float*)(ws + 4882432);     //  65536 B   (end ~4.72 MiB)

  k_prep_w1<<<DIN, DH, 0, stream>>>(W1, w1t_hi, w1t_lo);
  k_rnorm  <<<(G_ * N_) / 8, 256, 0, stream>>>(x, rnorm);
  k_csum   <<<G_, DIN, 0, stream>>>(x, rnorm, c);
  k_wcoef  <<<(G_ * N_) / 8, 256, 0, stream>>>(x, rnorm, c, w);
  k_fc1_pool<<<(G_ * N_ / 16) / 8, 256, 0, stream>>>(x, w1t_hi, w1t_lo, b1, w, partial);
  k_reduce <<<G_, DH, 0, stream>>>(partial, pooled);
  k_head   <<<G_, 32, 0, stream>>>(pooled, W2, b2, out);
}
